// PNATransformerEncoder_54443005444580
// MI455X (gfx1250) — matrix operand order repaired and checked by reading, not compiled, not run on hardware
//
#include <hip/hip_runtime.h>
#include <hip/hip_bf16.h>
#include <math.h>

static constexpr int kD    = 128;
static constexpr int kT    = 4;
static constexpr int kFIN  = 32;
static constexpr int kDMLP = 512;
static constexpr int kL    = 2;
#define AVG_LOG_F 2.8332133440562162f

typedef __bf16 bf16;
typedef __attribute__((ext_vector_type(16))) __bf16 v16bf;
typedef __attribute__((ext_vector_type(8)))  float  v8f;

union U16frag { uint4 q[2]; v16bf v; };

__device__ inline v8f vzero8() {
    v8f c;
#pragma unroll
    for (int i = 0; i < 8; ++i) c[i] = 0.f;
    return c;
}

__device__ inline v8f wmma_bf16(v16bf a, v16bf b, v8f c) {
    return __builtin_amdgcn_wmma_f32_16x16x32_bf16(
        false, a, false, b, (short)0, c, false, false);
}

// A-fragment element position for 16-bit A (16x32), per ISA layout
__device__ inline int a_pos(int khalf, int j) {
    return (j < 8) ? (khalf * 8 + j) : (16 + khalf * 8 + (j - 8));
}

// Packed A-load: row of 32 contiguous bf16 (16B-aligned). Two b128 loads.
__device__ inline v16bf load_a_packed(const bf16* rowp, int khalf) {
    U16frag u;
    u.q[0] = *reinterpret_cast<const uint4*>(rowp + khalf * 8);
    u.q[1] = *reinterpret_cast<const uint4*>(rowp + 16 + khalf * 8);
    return u.v;
}

// Packed B-load from pre-swizzled fragment buffer: lane's 16 bf16 contiguous.
__device__ inline v16bf load_b_packed(const bf16* F, int NT, int kb, int nt, int lane) {
    const uint4* p = reinterpret_cast<const uint4*>(F + ((((size_t)kb * NT + nt) * 32 + lane) << 4));
    U16frag u;
    u.q[0] = p[0];
    u.q[1] = p[1];
    return u.v;
}

// Order-preserving float <-> uint mapping for atomic min/max
__device__ inline unsigned fkey(float x) {
    unsigned u = __float_as_uint(x);
    return (u & 0x80000000u) ? ~u : (u | 0x80000000u);
}
__device__ inline float funkey(unsigned k) {
    unsigned u = (k & 0x80000000u) ? (k & 0x7fffffffu) : ~k;
    return __uint_as_float(u);
}

// ---------------------------------------------------------------- utility
__global__ void k_copy_f32(float* __restrict__ o, const float* __restrict__ i, int n) {
    int t = blockIdx.x * blockDim.x + threadIdx.x;
    if (t < n) o[t] = i[t];
}
__global__ void k_zero_f32(float* __restrict__ o, int n) {
    int t = blockIdx.x * blockDim.x + threadIdx.x;
    if (t < n) o[t] = 0.f;
}
__global__ void k_zero_agg(float* __restrict__ s, float* __restrict__ s2,
                           unsigned* __restrict__ mn, unsigned* __restrict__ mx, int n) {
    int t = blockIdx.x * blockDim.x + threadIdx.x;
    if (t < n) { s[t] = 0.f; s2[t] = 0.f; mn[t] = 0xFFFFFFFFu; mx[t] = 0u; }
}
__global__ void k_count(float* __restrict__ cnt, const int* __restrict__ dst, int e) {
    int t = blockIdx.x * blockDim.x + threadIdx.x;
    if (t < e) atomicAdd(cnt + dst[t], 1.0f);
}
__global__ void k_nodeparams(const float* __restrict__ cnt, float* __restrict__ degv,
                             float* __restrict__ ampv, float* __restrict__ attv,
                             float* __restrict__ hasv, int n) {
    int t = blockIdx.x * blockDim.x + threadIdx.x;
    if (t < n) {
        float c = cnt[t];
        float deg = fmaxf(c, 1.0f);
        float lg = logf(deg + 1.0f);
        degv[t] = deg;
        ampv[t] = lg / AVG_LOG_F;
        attv[t] = AVG_LOG_F / lg;
        hasv[t] = (c > 0.f) ? 1.f : 0.f;
    }
}

// Pack batch of row-major KxNcols f32 matrices into bf16 WMMA B-fragment order:
// o[(((kb*NT+nt)*32+lane)*16)+j] = W[kb*32 + (j<8 ? (lane>>4)*8 + j : 16 + (lane>>4)*8 + j-8)][nt*16 + (lane&15)]
__global__ void k_pack(bf16* __restrict__ o, const float* __restrict__ W,
                       int K, int Ncols, int batch) {
    int t = blockIdx.x * blockDim.x + threadIdx.x;
    int per = K * Ncols;
    if (t >= per * batch) return;
    int bidx = t / per;
    int r = t - bidx * per;
    int j    = r & 15;
    int lane = (r >> 4) & 31;
    int fi   = r >> 9;              // kb*NT + nt
    int NT   = Ncols >> 4;
    int nt   = fi % NT;
    int kb   = fi / NT;
    int k = kb * 32 + ((j < 8) ? ((lane >> 4) * 8 + j) : (16 + (lane >> 4) * 8 + (j - 8)));
    int n = nt * 16 + (lane & 15);
    o[t] = (bf16)W[(size_t)bidx * per + (size_t)k * Ncols + n];
}

// ------------------------------------------------------- LayerNorm -> bf16
__global__ void k_ln1(const float* __restrict__ x, const float* __restrict__ g,
                      const float* __restrict__ b, bf16* __restrict__ h, int n) {
    int wid  = (blockIdx.x * blockDim.x + threadIdx.x) >> 5;
    int lane = threadIdx.x & 31;
    if (wid >= n) return;
    const float* xr = x + (size_t)wid * kD;
    float v0 = xr[lane * 4 + 0], v1 = xr[lane * 4 + 1];
    float v2 = xr[lane * 4 + 2], v3 = xr[lane * 4 + 3];
    float s = v0 + v1 + v2 + v3;
#pragma unroll
    for (int o2 = 16; o2 > 0; o2 >>= 1) s += __shfl_xor(s, o2, 32);
    float mu = s * (1.f / kD);
    float d0 = v0 - mu, d1 = v1 - mu, d2 = v2 - mu, d3 = v3 - mu;
    float q = d0 * d0 + d1 * d1 + d2 * d2 + d3 * d3;
#pragma unroll
    for (int o2 = 16; o2 > 0; o2 >>= 1) q += __shfl_xor(q, o2, 32);
    float rsig = rsqrtf(q * (1.f / kD) + 1e-5f);
    bf16* hr = h + (size_t)wid * kD;
#pragma unroll
    for (int i = 0; i < 4; ++i) {
        int c = lane * 4 + i;
        float dv = (i == 0 ? d0 : i == 1 ? d1 : i == 2 ? d2 : d3);
        hr[c] = (bf16)(dv * rsig * g[c] + b[c]);
    }
}

// ---------------------------------------------- fused edge GEMM + aggregate
// 4 waves/block, 16 edges/wave
__global__ void k_edge(const bf16* __restrict__ h, const int* __restrict__ src,
                       const int* __restrict__ dst, const bf16* __restrict__ WpreF,
                       const float* __restrict__ bpre,
                       float* __restrict__ sumb, float* __restrict__ sumsq,
                       unsigned* __restrict__ mink, unsigned* __restrict__ maxk, int E) {
    int lane  = threadIdx.x & 31;
    int wave  = threadIdx.x >> 5;
    int e0    = (blockIdx.x * 4 + wave) * 16;
    if (e0 >= E) return;
    int row   = lane & 15;
    int khalf = lane >> 4;

    int er = e0 + row < E ? e0 + row : E - 1;
    int nd = dst[er];
    int ns = src[er];

    int dn[8];
#pragma unroll
    for (int r = 0; r < 8; ++r) {
        int e = e0 + khalf * 8 + r;
        dn[r] = dst[e < E ? e : E - 1];
    }

#pragma unroll
    for (int t = 0; t < kT; ++t) {
        v8f acc[2];
        acc[0] = vzero8(); acc[1] = vzero8();
        const bf16* wb = WpreF + (size_t)t * (2 * kFIN * kFIN);  // fragment-packed 64x32, NT=2
#pragma unroll
        for (int kb = 0; kb < 2; ++kb) {
            int node = (kb == 0) ? nd : ns;
            v16bf a = load_a_packed(h + (size_t)node * kD + t * kFIN, khalf);
            acc[0] = wmma_bf16(a, load_b_packed(wb, 2, kb, 0, lane), acc[0]);
            acc[1] = wmma_bf16(a, load_b_packed(wb, 2, kb, 1, lane), acc[1]);
        }
#pragma unroll
        for (int nt = 0; nt < 2; ++nt) {
            int colT = nt * 16 + row;
            float bias = bpre[t * kFIN + colT];
            int col = t * kFIN + colT;
#pragma unroll
            for (int r = 0; r < 8; ++r) {
                int e = e0 + khalf * 8 + r;
                if (e < E) {
                    float v = acc[nt][r] + bias;
                    size_t idx = (size_t)dn[r] * kD + col;
                    atomicAdd(sumb + idx, v);
                    atomicAdd(sumsq + idx, v * v);
                    atomicMin(mink + idx, fkey(v));
                    atomicMax(maxk + idx, fkey(v));
                }
            }
        }
    }
}

// --------------------------------- per-node: comb @ W_post -> @ W_lin + res
// 4 waves/block, 16 nodes/wave
__global__ void k_post(float* __restrict__ x, const bf16* __restrict__ h,
                       const float* __restrict__ sumb, const float* __restrict__ sumsq,
                       const unsigned* __restrict__ mink, const unsigned* __restrict__ maxk,
                       const float* __restrict__ degv, const float* __restrict__ ampv,
                       const float* __restrict__ attv, const float* __restrict__ hasv,
                       const bf16* __restrict__ WpostF, const float* __restrict__ bpost,
                       const bf16* __restrict__ WlinF, const float* __restrict__ blin,
                       int N) {
    __shared__ alignas(16) bf16 o_bf[4][16][kD];
    int lane  = threadIdx.x & 31;
    int wave  = threadIdx.x >> 5;
    int n0    = (blockIdx.x * 4 + wave) * 16;
    if (n0 >= N) return;
    int row   = lane & 15;
    int khalf = lane >> 4;

    int nA = n0 + row < N ? n0 + row : N - 1;
    float rdeg = 1.f / degv[nA];
    float amp = ampv[nA], att = attv[nA], has = hasv[nA];

    for (int t = 0; t < kT; ++t) {
        v8f acc[2];
        acc[0] = vzero8(); acc[1] = vzero8();
        const bf16* wb = WpostF + (size_t)t * (13 * kFIN * kFIN);  // packed 416x32, NT=2
        // kb == 0: pure h block (packed load)
        {
            v16bf a = load_a_packed(h + (size_t)nA * kD + t * kFIN, khalf);
            acc[0] = wmma_bf16(a, load_b_packed(wb, 2, 0, 0, lane), acc[0]);
            acc[1] = wmma_bf16(a, load_b_packed(wb, 2, 0, 1, lane), acc[1]);
        }
        // kb >= 1: computed aggregation features
        for (int kb = 1; kb < 13; ++kb) {
            v16bf a;
#pragma unroll
            for (int j = 0; j < 16; ++j) {
                int jj = (kb - 1) * 32 + a_pos(khalf, j);   // 0..383
                int g  = jj >> 7;                            // 0 raw, 1 amp, 2 att
                int a2 = jj & 127;
                int which = a2 >> 5;                         // mean,min,max,std
                int e = a2 & 31;
                size_t idx = (size_t)nA * kD + t * kFIN + e;
                float mean = sumb[idx] * rdeg;
                float v;
                if (which == 0)      v = mean;
                else if (which == 1) v = (has != 0.f) ? funkey(mink[idx]) : 0.f;
                else if (which == 2) v = (has != 0.f) ? funkey(maxk[idx]) : 0.f;
                else {
                    float m2 = sumsq[idx] * rdeg;
                    v = sqrtf(fmaxf(m2 - mean * mean, 0.f) + 1e-5f);
                }
                a[j] = (bf16)(v * (g == 0 ? 1.f : (g == 1 ? amp : att)));
            }
            acc[0] = wmma_bf16(a, load_b_packed(wb, 2, kb, 0, lane), acc[0]);
            acc[1] = wmma_bf16(a, load_b_packed(wb, 2, kb, 1, lane), acc[1]);
        }
#pragma unroll
        for (int nt = 0; nt < 2; ++nt) {
            int colT = nt * 16 + row;
#pragma unroll
            for (int r = 0; r < 8; ++r)
                o_bf[wave][khalf * 8 + r][t * kFIN + colT] =
                    (bf16)(acc[nt][r] + bpost[t * kFIN + colT]);
        }
    }
    __syncthreads();

    // o (16x128) @ W_lin (128x128) + b_lin + x
    v16bf afr[4];
#pragma unroll
    for (int kb = 0; kb < 4; ++kb)
        afr[kb] = load_a_packed(&o_bf[wave][row][kb * 32], khalf);

    for (int nt = 0; nt < 8; ++nt) {
        v8f c = vzero8();
#pragma unroll
        for (int kb = 0; kb < 4; ++kb)
            c = wmma_bf16(afr[kb], load_b_packed(WlinF, 8, kb, nt, lane), c);
        int col = nt * 16 + row;
#pragma unroll
        for (int r = 0; r < 8; ++r) {
            int node = n0 + khalf * 8 + r;
            if (node < N) {
                size_t idx = (size_t)node * kD + col;
                x[idx] = x[idx] + c[r] + blin[col];
            }
        }
    }
}

// -------------------------------------------- LN2 + MLP (128->512->128) + res
// 1 wave/block, 16 nodes
__global__ void k_mlp(float* __restrict__ x, const float* __restrict__ g2,
                      const float* __restrict__ bb2, const bf16* __restrict__ W1F,
                      const float* __restrict__ b1, const bf16* __restrict__ W2F,
                      const float* __restrict__ b2, int N) {
    __shared__ alignas(16) float xs[16][kD];
    __shared__ alignas(16) bf16 h2[16][kD];
    __shared__ alignas(16) bf16 hid[16][kDMLP];
    __shared__ float mu[16], rs[16];
    int lane  = threadIdx.x & 31;
    int n0    = blockIdx.x * 16;
    int row   = lane & 15;
    int khalf = lane >> 4;

    for (int i = lane; i < 16 * kD; i += 32) {
        int r = i >> 7, c = i & 127;
        int node = n0 + r < N ? n0 + r : N - 1;
        xs[r][c] = x[(size_t)node * kD + c];
    }
    __syncthreads();
    if (lane < 16) {
        float s = 0.f;
        for (int j = 0; j < kD; ++j) s += xs[lane][j];
        float m = s * (1.f / kD);
        float q = 0.f;
        for (int j = 0; j < kD; ++j) { float d = xs[lane][j] - m; q += d * d; }
        mu[lane] = m;
        rs[lane] = rsqrtf(q * (1.f / kD) + 1e-5f);
    }
    __syncthreads();
    for (int i = lane; i < 16 * kD; i += 32) {
        int r = i >> 7, c = i & 127;
        h2[r][c] = (bf16)((xs[r][c] - mu[r]) * rs[r] * g2[c] + bb2[c]);
    }
    __syncthreads();

    v16bf afr[4];
#pragma unroll
    for (int kb = 0; kb < 4; ++kb)
        afr[kb] = load_a_packed(&h2[row][kb * 32], khalf);

    // h2 (16x128) @ W1 (128x512), exact GELU, -> hid bf16
    for (int nt = 0; nt < 32; ++nt) {
        v8f c = vzero8();
#pragma unroll
        for (int kb = 0; kb < 4; ++kb)
            c = wmma_bf16(afr[kb], load_b_packed(W1F, 32, kb, nt, lane), c);
        int col = nt * 16 + row;
#pragma unroll
        for (int r = 0; r < 8; ++r) {
            int m = khalf * 8 + r;
            float z = c[r] + b1[col];
            float ge = 0.5f * z * (1.f + erff(z * 0.70710678118654752f));
            hid[m][col] = (bf16)ge;
        }
    }
    __syncthreads();

    // hid (16x512) @ W2 (512x128) + b2 + residual
    for (int nt = 0; nt < 8; ++nt) {
        v8f c = vzero8();
        for (int kb = 0; kb < 16; ++kb) {
            v16bf a = load_a_packed(&hid[row][kb * 32], khalf);
            c = wmma_bf16(a, load_b_packed(W2F, 8, kb, nt, lane), c);
        }
        int col = nt * 16 + row;
#pragma unroll
        for (int r = 0; r < 8; ++r) {
            int m = khalf * 8 + r;
            int node = n0 + m;
            if (node < N)
                x[(size_t)node * kD + col] = xs[m][col] + c[r] + b2[col];
        }
    }
}

// ---------------------------------------------------------------------------
extern "C" void kernel_launch(void* const* d_in, const int* in_sizes, int n_in,
                              void* d_out, int out_size, void* d_ws, size_t ws_size,
                              hipStream_t stream) {
    const float* x_in   = (const float*)d_in[0];
    const int*   ei     = (const int*)d_in[1];
    const float* ln1_g  = (const float*)d_in[2];
    const float* ln1_b  = (const float*)d_in[3];
    const float* W_pre  = (const float*)d_in[4];
    const float* b_pre  = (const float*)d_in[5];
    const float* W_post = (const float*)d_in[6];
    const float* b_post = (const float*)d_in[7];
    const float* W_lin  = (const float*)d_in[8];
    const float* b_lin  = (const float*)d_in[9];
    const float* ln2_g  = (const float*)d_in[10];
    const float* ln2_b  = (const float*)d_in[11];
    const float* W1     = (const float*)d_in[12];
    const float* b1     = (const float*)d_in[13];
    const float* W2     = (const float*)d_in[14];
    const float* b2     = (const float*)d_in[15];

    const int N = in_sizes[0] / kD;
    const int E = in_sizes[1] / 2;
    const int* src = ei;
    const int* dst = ei + E;

    char* ws = (char*)d_ws;
    size_t off = 0;
    auto take = [&](size_t bytes) -> char* {
        char* p = ws + off;
        off = (off + bytes + 255) & ~(size_t)255;
        return p;
    };
    float*    x_cur = (float*)take((size_t)N * kD * 4);
    bf16*     h_bf  = (bf16*)take((size_t)N * kD * 2);
    float*    sumb  = (float*)take((size_t)N * kD * 4);
    float*    sumsq = (float*)take((size_t)N * kD * 4);
    unsigned* mink  = (unsigned*)take((size_t)N * kD * 4);
    unsigned* maxk  = (unsigned*)take((size_t)N * kD * 4);
    float*    cnt   = (float*)take((size_t)N * 4);
    float*    degv  = (float*)take((size_t)N * 4);
    float*    ampv  = (float*)take((size_t)N * 4);
    float*    attv  = (float*)take((size_t)N * 4);
    float*    hasv  = (float*)take((size_t)N * 4);
    const int nWpre  = kL * kT * 2 * kFIN * kFIN;
    const int nWpost = kL * kT * 13 * kFIN * kFIN;
    const int nWlin  = kL * kD * kD;
    const int nW1    = kL * kD * kDMLP;
    const int nW2    = kL * kDMLP * kD;
    bf16* WpreF  = (bf16*)take((size_t)nWpre * 2);
    bf16* WpostF = (bf16*)take((size_t)nWpost * 2);
    bf16* WlinF  = (bf16*)take((size_t)nWlin * 2);
    bf16* W1F    = (bf16*)take((size_t)nW1 * 2);
    bf16* W2F    = (bf16*)take((size_t)nW2 * 2);
    (void)ws_size; (void)n_in;

    const int ND = N * kD;
    k_copy_f32<<<(ND + 255) / 256, 256, 0, stream>>>(x_cur, x_in, ND);
    k_zero_f32<<<(N + 255) / 256, 256, 0, stream>>>(cnt, N);
    k_count<<<(E + 255) / 256, 256, 0, stream>>>(cnt, dst, E);
    k_nodeparams<<<(N + 255) / 256, 256, 0, stream>>>(cnt, degv, ampv, attv, hasv, N);
    // pack weights into WMMA fragment order (bf16)
    k_pack<<<(nWpre + 255) / 256, 256, 0, stream>>>(WpreF, W_pre, 2 * kFIN, kFIN, kL * kT);
    k_pack<<<(nWpost + 255) / 256, 256, 0, stream>>>(WpostF, W_post, 13 * kFIN, kFIN, kL * kT);
    k_pack<<<(nWlin + 255) / 256, 256, 0, stream>>>(WlinF, W_lin, kD, kD, kL);
    k_pack<<<(nW1 + 255) / 256, 256, 0, stream>>>(W1F, W1, kD, kDMLP, kL);
    k_pack<<<(nW2 + 255) / 256, 256, 0, stream>>>(W2F, W2, kDMLP, kD, kL);

    for (int l = 0; l < kL; ++l) {
        k_zero_agg<<<(ND + 255) / 256, 256, 0, stream>>>(sumb, sumsq, mink, maxk, ND);
        k_ln1<<<(N + 7) / 8, 256, 0, stream>>>(x_cur, ln1_g + l * kD, ln1_b + l * kD, h_bf, N);
        int ewaves = (E + 15) / 16;
        k_edge<<<(ewaves + 3) / 4, 128, 0, stream>>>(
            h_bf, src, dst,
            WpreF + (size_t)l * kT * 2 * kFIN * kFIN, b_pre + l * kT * kFIN,
            sumb, sumsq, mink, maxk, E);
        int nwaves = (N + 15) / 16;
        k_post<<<(nwaves + 3) / 4, 128, 0, stream>>>(
            x_cur, h_bf, sumb, sumsq, mink, maxk, degv, ampv, attv, hasv,
            WpostF + (size_t)l * kT * 13 * kFIN * kFIN, b_post + l * kT * kFIN,
            WlinF + (size_t)l * kD * kD, b_lin + l * kD, N);
        k_mlp<<<nwaves, 32, 0, stream>>>(
            x_cur, ln2_g + l * kD, ln2_b + l * kD,
            W1F + (size_t)l * kD * kDMLP, b1 + l * kDMLP,
            W2F + (size_t)l * kDMLP * kD, b2 + l * kD, N);
    }
    k_copy_f32<<<(out_size + 255) / 256, 256, 0, stream>>>((float*)d_out, x_cur, out_size);
}
